// SampleNetDD_55731495633215
// MI455X (gfx1250) — compile-verified, matrix operand order checked by reading, not run on hardware
//
#include <hip/hip_runtime.h>

#define BB   16
#define NN   1024
#define KNB  20
#define LL   7
#define KS   8
#define FF   8
#define CC   40
#define HH   256
#define INDD 27
#define KPAD 32
#define NPTS (BB*NN)   // 16384

typedef __attribute__((ext_vector_type(16))) _Float16 v16h;
typedef __attribute__((ext_vector_type(8)))  float    v8f;

union V16H { v16h v; _Float16 e[16]; };
union V8F  { v8f  v; float    e[8];  };

// K-index mapping within a 16x32 fp16 WMMA A tile (mirrored for B):
// lanes 0-15 hold K {0-7,16-23}, lanes 16-31 hold K {8-15,24-31}.
__device__ __forceinline__ int kmap(int lane, int h) {
    int lo = h & 7, hi = h >> 3;
    return (lane < 16) ? (hi ? 16 + lo : lo) : (hi ? 24 + lo : 8 + lo);
}
// Inverse: element (m_local, k_local) of a 16x32 tile lives at (lane, half):
__device__ __forceinline__ int inv_lane(int m_local, int k_local) {
    return m_local + (((k_local & 15) >= 8) ? 16 : 0);
}
__device__ __forceinline__ int inv_half(int k_local) {
    return ((k_local >= 16) ? 8 : 0) + (k_local & 7);
}

// ---------------------------------------------------------------- KNN
__global__ __launch_bounds__(128)
void knn_kernel(const float* __restrict__ pos, int* __restrict__ idx,
                float* __restrict__ offs) {
    __shared__ float px[NN], py[NN], pz[NN];
    __shared__ float dist[NN];
    __shared__ float rd[128];
    __shared__ int   ri[128];
    int point = blockIdx.x;           // 0..16383
    int b = point >> 10, n = point & 1023;
    int tid = threadIdx.x;
    for (int i = tid; i < NN; i += 128) {
        const float* p = pos + (size_t)(b * NN + i) * 3;
        px[i] = p[0]; py[i] = p[1]; pz[i] = p[2];
    }
    __syncthreads();
    float qx = px[n], qy = py[n], qz = pz[n];
    for (int i = tid; i < NN; i += 128) {
        float dx = px[i]-qx, dy = py[i]-qy, dz = pz[i]-qz;
        float d = dx*dx + dy*dy + dz*dz;
        if (i == n) d += 1e10f;       // diagonal mask like reference
        dist[i] = d;
    }
    __syncthreads();
    for (int k = 0; k < KNB; ++k) {
        float bd = 3.0e38f; int bi = NN;
        for (int i = tid; i < NN; i += 128) {
            float d = dist[i];
            if (d < bd || (d == bd && i < bi)) { bd = d; bi = i; }
        }
        rd[tid] = bd; ri[tid] = bi;
        __syncthreads();
        for (int s = 64; s > 0; s >>= 1) {
            if (tid < s) {
                float d2 = rd[tid+s]; int i2 = ri[tid+s];
                if (d2 < rd[tid] || (d2 == rd[tid] && i2 < ri[tid])) {
                    rd[tid] = d2; ri[tid] = i2;
                }
            }
            __syncthreads();
        }
        if (tid == 0) {
            int bi0 = ri[0];
            dist[bi0] = 3.0e38f;
            idx[(size_t)point * KNB + k] = bi0;
            float* o = offs + ((size_t)point * KNB + k) * 3;
            o[0] = px[bi0]-qx; o[1] = py[bi0]-qy; o[2] = pz[bi0]-qz;
        }
        __syncthreads();
    }
}

// ---------------------------------------------------------------- 3x3 eigh
__device__ void eigvec3(float a00,float a01,float a02,float a11,float a12,float a22,
                        float lam, float v[3]) {
    float r0x=a00-lam, r0y=a01,     r0z=a02;
    float r1x=a01,     r1y=a11-lam, r1z=a12;
    float r2x=a02,     r2y=a12,     r2z=a22-lam;
    float c0x=r0y*r1z-r0z*r1y, c0y=r0z*r1x-r0x*r1z, c0z=r0x*r1y-r0y*r1x;
    float c1x=r0y*r2z-r0z*r2y, c1y=r0z*r2x-r0x*r2z, c1z=r0x*r2y-r0y*r2x;
    float c2x=r1y*r2z-r1z*r2y, c2y=r1z*r2x-r1x*r2z, c2z=r1x*r2y-r1y*r2x;
    float n0=c0x*c0x+c0y*c0y+c0z*c0z;
    float n1=c1x*c1x+c1y*c1y+c1z*c1z;
    float n2=c2x*c2x+c2y*c2y+c2z*c2z;
    float bx=c0x, by=c0y, bz=c0z, bn=n0;
    if (n1 > bn) { bx=c1x; by=c1y; bz=c1z; bn=n1; }
    if (n2 > bn) { bx=c2x; by=c2y; bz=c2z; bn=n2; }
    if (bn < 1e-24f) { v[0]=1.0f; v[1]=0.0f; v[2]=0.0f; return; }
    float inv = rsqrtf(bn);
    v[0]=bx*inv; v[1]=by*inv; v[2]=bz*inv;
}

__device__ void eigh3(float a00,float a01,float a02,float a11,float a12,float a22,
                      float E[3][3]) {   // columns = eigenvectors, ascending eigvals
    float p1 = a01*a01 + a02*a02 + a12*a12;
    float q  = (a00+a11+a22) * (1.0f/3.0f);
    float d0 = a00-q, d1 = a11-q, d2 = a22-q;
    float p2 = d0*d0 + d1*d1 + d2*d2 + 2.0f*p1;
    if (p2 < 1e-20f) {
        E[0][0]=1;E[0][1]=0;E[0][2]=0;
        E[1][0]=0;E[1][1]=1;E[1][2]=0;
        E[2][0]=0;E[2][1]=0;E[2][2]=1;
        return;
    }
    float p  = sqrtf(p2 * (1.0f/6.0f));
    float ip = 1.0f / p;
    float b00=d0*ip, b01=a01*ip, b02=a02*ip, b11=d1*ip, b12=a12*ip, b22=d2*ip;
    float detB = b00*(b11*b22-b12*b12) - b01*(b01*b22-b12*b02) + b02*(b01*b12-b11*b02);
    float r = fminf(1.0f, fmaxf(-1.0f, 0.5f*detB));
    float phi = acosf(r) * (1.0f/3.0f);
    float l2 = q + 2.0f*p*cosf(phi);                       // largest
    float l0 = q + 2.0f*p*cosf(phi + 2.0943951023931953f); // smallest
    float v0[3], v2[3];
    eigvec3(a00,a01,a02,a11,a12,a22, l0, v0);
    eigvec3(a00,a01,a02,a11,a12,a22, l2, v2);
    float v1x = v2[1]*v0[2]-v2[2]*v0[1];
    float v1y = v2[2]*v0[0]-v2[0]*v0[2];
    float v1z = v2[0]*v0[1]-v2[1]*v0[0];
    float nn = v1x*v1x+v1y*v1y+v1z*v1z;
    float inv = (nn > 1e-24f) ? rsqrtf(nn) : 0.0f;
    v1x*=inv; v1y*=inv; v1z*=inv;
    E[0][0]=v0[0]; E[1][0]=v0[1]; E[2][0]=v0[2];
    E[0][1]=v1x;   E[1][1]=v1y;   E[2][1]=v1z;
    E[0][2]=v2[0]; E[1][2]=v2[1]; E[2][2]=v2[2];
}

// ---------------------------------------------------------------- geometry + spline features
__global__ __launch_bounds__(256)
void geom_kernel(const float* __restrict__ offs, const float* __restrict__ wsp,
                 float* __restrict__ proj, float* __restrict__ feat) {
    int t = blockIdx.x * 256 + threadIdx.x;
    if (t >= NPTS) return;
    float ox[KNB], oy[KNB], oz[KNB];
    const float* o = offs + (size_t)t * KNB * 3;
    for (int k = 0; k < KNB; ++k) { ox[k]=o[k*3+0]; oy[k]=o[k*3+1]; oz[k]=o[k*3+2]; }
    float c00=0,c01=0,c02=0,c11=0,c12=0,c22=0;
    for (int k = 0; k < LL; ++k) {
        c00 += ox[k]*ox[k]; c01 += ox[k]*oy[k]; c02 += ox[k]*oz[k];
        c11 += oy[k]*oy[k]; c12 += oy[k]*oz[k]; c22 += oz[k]*oz[k];
    }
    float s7 = 1.0f / (float)LL;
    c00*=s7; c01*=s7; c02*=s7; c11*=s7; c12*=s7; c22*=s7;
    float E[3][3];
    eigh3(c00,c01,c02,c11,c12,c22,E);
    float pj[KNB][3];
    float sc[3] = {0.0f, 0.0f, 0.0f};
    for (int k = 0; k < KNB; ++k)
        for (int e = 0; e < 3; ++e) {
            float v = ox[k]*E[0][e] + oy[k]*E[1][e] + oz[k]*E[2][e];
            pj[k][e] = v;
            sc[e] = fmaxf(sc[e], fabsf(v));
        }
    for (int e = 0; e < 3; ++e) sc[e] += 1e-8f;
    float fsum[3][FF] = {};
    for (int k = 0; k < KNB; ++k)
        for (int d = 0; d < 3; ++d) {
            float u = (pj[k][d] / sc[d] + 1.0f) * 0.5f;
            u = fminf(1.0f, fmaxf(0.0f, u));
            float tt = u * (float)(KS - 1);
            int i0 = (int)floorf(tt);
            i0 = i0 < 0 ? 0 : (i0 > KS-2 ? KS-2 : i0);
            float fr = tt - (float)i0;
            const float* w0 = wsp + (d*KS + i0) * FF;
            for (int f = 0; f < FF; ++f)
                fsum[d][f] += (1.0f - fr) * w0[f] + fr * w0[FF + f];
        }
    float* pr = proj + (size_t)t * KNB * 3;
    for (int k = 0; k < KNB; ++k)
        for (int e = 0; e < 3; ++e) pr[k*3+e] = pj[k][e];
    float* ft = feat + (size_t)t * 24;
    float invK = 1.0f / (float)KNB;
    for (int d = 0; d < 3; ++d)
        for (int f = 0; f < FF; ++f) {
            float x = fsum[d][f] * invK;
            ft[d*FF + f] = 1.0f / (1.0f + expf(-x));   // sigmoid
        }
}

// ---------------------------------------------------------------- weight repack to WMMA-B lane layout
__global__ __launch_bounds__(256)
void pack_kernel(const float* __restrict__ w1, const float* __restrict__ w2,
                 const float* __restrict__ w3, _Float16* __restrict__ w1p,
                 _Float16* __restrict__ w2p, _Float16* __restrict__ w3p) {
    const int total1 = 16*32*16;      // 8192  (1 kt x 16 nt)
    const int total2 = 8*16*32*16;    // 65536 (8 kt x 16 nt)
    int t = blockIdx.x * 256 + threadIdx.x;
    if (t < total1) {
        int h = t & 15, lane = (t >> 4) & 31, nt = t >> 9;
        int n = nt*16 + (lane & 15);
        int k = kmap(lane, h);
        float v = (k < INDD) ? w1[k*HH + n] : 0.0f;
        w1p[t] = (_Float16)v;
    } else if (t < total1 + total2) {
        int e = t - total1;
        int h = e & 15, lane = (e >> 4) & 31, nt = (e >> 9) & 15, kt = e >> 13;
        int n = nt*16 + (lane & 15);
        int k = kt*32 + kmap(lane, h);
        w2p[e] = (_Float16)w2[k*HH + n];
    } else if (t < total1 + 2*total2) {
        int e = t - total1 - total2;
        int h = e & 15, lane = (e >> 4) & 31, nt = (e >> 9) & 15, kt = e >> 13;
        int n = nt*16 + (lane & 15);
        int k = kt*32 + kmap(lane, h);
        w3p[e] = (_Float16)w3[k*HH + n];
    }
}

// ---------------------------------------------------------------- fused edge MLP + max-pool
// 4 nodes (=80 edges) per WG; activations live in LDS in WMMA-A register layout:
// A1p: [mt=5][lane=32][h=16]  H1p: [kt=8][mt=5][lane=32][h=16]
__global__ __launch_bounds__(256)
void mlp_kernel(const float* __restrict__ proj, const float* __restrict__ feat,
                const int* __restrict__ idx, const _Float16* __restrict__ w1p,
                const float* __restrict__ b1, const _Float16* __restrict__ w2p,
                const float* __restrict__ b2, float* __restrict__ nodebuf) {
    __shared__ __attribute__((aligned(32))) _Float16 A1p[5 * 32 * 16];      //  5 KB
    __shared__ __attribute__((aligned(32))) _Float16 H1p[8 * 5 * 32 * 16];  // 40 KB
    __shared__ unsigned nmax[4 * HH];                                        //  4 KB
    int tid  = threadIdx.x;
    int lane = tid & 31, wave = tid >> 5;
    int g0   = blockIdx.x * 4;           // first node of this WG
    for (int i = tid; i < 4*HH; i += 256) nmax[i] = 0u;
    // Build edge_in = [proj(3) | feat_j(24) | 0 pad] directly into A-packed layout
    for (int i = tid; i < 80*KPAD; i += 256) {
        int row = i >> 5, c = i & 31;    // c = k_local within the single K-tile
        int nl = row / KNB, kk = row % KNB;
        int node = g0 + nl;
        float v = 0.0f;
        if (c < 3) {
            v = proj[((size_t)node*KNB + kk)*3 + c];
        } else if (c < INDD) {
            int j = idx[(size_t)node*KNB + kk];
            int b = node >> 10;
            v = feat[((size_t)b*NN + j)*24 + (c - 3)];
        }
        int mt = row >> 4, m_local = row & 15;
        A1p[(mt*32 + inv_lane(m_local, c))*16 + inv_half(c)] = (_Float16)v;
    }
    __syncthreads();
    const v16h* w1v = (const v16h*)w1p;
    const v16h* w2v = (const v16h*)w2p;
    const v16h* a1v = (const v16h*)A1p;
    const v16h* h1v = (const v16h*)H1p;
    int m = lane & 15;
    int rsel = (lane < 16) ? 0 : 8;
    // ---- layer 1: 5 mt x 16 nt tiles, K=32 (one WMMA each)
    for (int t = wave*10; t < wave*10 + 10; ++t) {
        int mt = t >> 4, nt = t & 15;
        V16H a; a.v = a1v[mt*32 + lane];           // 2x ds_load_b128
        v16h bm = w1v[nt*32 + lane];               // 2x global_load_b128
        v8f c = {};
        c = __builtin_amdgcn_wmma_f32_16x16x32_f16(false, a.v, false, bm,
                                                   (short)0, c, false, false);
        int col = nt*16 + m;
        float bias = b1[col];
        // scatter into H1p A-layout: kt = col/32, k_local = col%32
        int kt = nt >> 1;
        int hh = ((nt & 1) ? 8 : 0) + (m & 7);     // inv_half(col%32)
        int lt16 = (m >= 8) ? 16 : 0;              // (k_local&15)>=8  <=>  m>=8
        V8F cc; cc.v = c;
        #pragma unroll
        for (int r = 0; r < 8; ++r) {
            int row = mt*16 + r + rsel;
            int m_local = row & 15;
            float v = fmaxf(cc.e[r] + bias, 0.0f);
            H1p[((kt*5 + (row >> 4))*32 + m_local + lt16)*16 + hh] = (_Float16)v;
        }
    }
    __syncthreads();
    // ---- layer 2: same 80 tiles, K=256 (8 chained WMMAs), relu + max over 20 edges
    for (int t = wave*10; t < wave*10 + 10; ++t) {
        int mt = t >> 4, nt = t & 15;
        v8f c = {};
        #pragma unroll
        for (int kt = 0; kt < 8; ++kt) {
            V16H a; a.v = h1v[(kt*5 + mt)*32 + lane];   // 2x ds_load_b128
            v16h bm = w2v[(kt*16 + nt)*32 + lane];
            c = __builtin_amdgcn_wmma_f32_16x16x32_f16(false, a.v, false, bm,
                                                       (short)0, c, false, false);
        }
        int col = nt*16 + m;
        float bias = b2[col];
        V8F cc; cc.v = c;
        #pragma unroll
        for (int r = 0; r < 8; ++r) {
            int row = mt*16 + r + rsel;
            float v = fmaxf(cc.e[r] + bias, 0.0f);   // relu => v >= 0, uint max == float max
            int nl = row / KNB;
            atomicMax(&nmax[nl*HH + col], __float_as_uint(v));
        }
    }
    __syncthreads();
    for (int i = tid; i < 4*HH; i += 256) {
        int nl = i >> 8, col = i & 255;
        nodebuf[(size_t)(g0 + nl)*HH + col] = __uint_as_float(nmax[i]);
    }
}

// ---------------------------------------------------------------- mean over N per (b,col)
__global__ __launch_bounds__(256)
void nodemean_kernel(const float* __restrict__ nodebuf, float* __restrict__ ysbuf) {
    __shared__ float red[256];
    int bb = blockIdx.x;                 // 0..BB*HH-1
    int b = bb >> 8, col = bb & 255;
    int tid = threadIdx.x;
    float s = 0.0f;
    for (int n = tid; n < NN; n += 256)
        s += nodebuf[((size_t)b*NN + n)*HH + col];
    red[tid] = s;
    __syncthreads();
    for (int st = 128; st > 0; st >>= 1) {
        if (tid < st) red[tid] += red[tid + st];
        __syncthreads();
    }
    if (tid == 0) ysbuf[bb] = red[0] * (1.0f / (float)NN);
}

// ---------------------------------------------------------------- head: WMMA fc3 + elu + fc4 + log_softmax
__global__ __launch_bounds__(256)
void head_kernel(const float* __restrict__ ysbuf, const _Float16* __restrict__ w3p,
                 const float* __restrict__ b3, const float* __restrict__ w4,
                 const float* __restrict__ b4, float* __restrict__ out) {
    __shared__ __attribute__((aligned(32))) _Float16 ysA[8 * 32 * 16]; // A-layout ys, 8 KB
    __shared__ float y1f[BB * HH];
    __shared__ float y2f[BB * CC];
    int tid = threadIdx.x;
    int lane = tid & 31, wave = tid >> 5;
    // pack ys (16x256 f32) into fp16 WMMA-A layout
    for (int i = tid; i < 8*32*16; i += 256) {
        int h = i & 15, ln = (i >> 4) & 31, kt = i >> 9;
        int mm = ln & 15;
        int k = kt*32 + kmap(ln, h);
        ysA[i] = (_Float16)ysbuf[mm*HH + k];
    }
    __syncthreads();
    const v16h* w3v = (const v16h*)w3p;
    const v16h* ysv = (const v16h*)ysA;
    int m = lane & 15;
    int rsel = (lane < 16) ? 0 : 8;
    for (int nt = wave*2; nt < wave*2 + 2; ++nt) {   // M=16 is exactly one M-tile
        v8f c = {};
        #pragma unroll
        for (int kt = 0; kt < 8; ++kt) {
            V16H a; a.v = ysv[kt*32 + lane];
            v16h bm = w3v[(kt*16 + nt)*32 + lane];
            c = __builtin_amdgcn_wmma_f32_16x16x32_f16(false, a.v, false, bm,
                                                       (short)0, c, false, false);
        }
        int col = nt*16 + m;
        float bias = b3[col];
        V8F cc; cc.v = c;
        #pragma unroll
        for (int r = 0; r < 8; ++r) {
            int row = r + rsel;
            float x = cc.e[r] + bias;
            y1f[row*HH + col] = (x > 0.0f) ? x : expm1f(x);   // elu
        }
    }
    __syncthreads();
    for (int o = tid; o < BB*CC; o += 256) {
        int row = o / CC, col = o % CC;
        float acc = b4[col];
        for (int k = 0; k < HH; ++k) acc += y1f[row*HH + k] * w4[k*CC + col];
        y2f[o] = acc;
    }
    __syncthreads();
    if (tid < BB) {
        int row = tid;
        float mx = -3.0e38f;
        for (int c = 0; c < CC; ++c) mx = fmaxf(mx, y2f[row*CC + c]);
        float s = 0.0f;
        for (int c = 0; c < CC; ++c) s += expf(y2f[row*CC + c] - mx);
        float lse = mx + logf(s);
        for (int c = 0; c < CC; ++c) out[row*CC + c] = y2f[row*CC + c] - lse;
    }
}

// ---------------------------------------------------------------- launch
extern "C" void kernel_launch(void* const* d_in, const int* in_sizes, int n_in,
                              void* d_out, int out_size, void* d_ws, size_t ws_size,
                              hipStream_t stream) {
    const float* pos = (const float*)d_in[0];
    const float* wsp = (const float*)d_in[1];
    const float* w1  = (const float*)d_in[2];
    const float* b1  = (const float*)d_in[3];
    const float* w2  = (const float*)d_in[4];
    const float* b2  = (const float*)d_in[5];
    const float* w3  = (const float*)d_in[6];
    const float* b3  = (const float*)d_in[7];
    const float* w4  = (const float*)d_in[8];
    const float* b4  = (const float*)d_in[9];
    float* out = (float*)d_out;
    (void)in_sizes; (void)n_in; (void)out_size; (void)ws_size;

    char* ws = (char*)d_ws;
    size_t off = 0;
    auto alloc = [&](size_t bytes) -> void* {
        void* p = ws + off;
        off = (off + bytes + 255) & ~(size_t)255;
        return p;
    };
    int*      idxb    = (int*)     alloc((size_t)NPTS*KNB*4);
    float*    offsb   = (float*)   alloc((size_t)NPTS*KNB*3*4);
    float*    projb   = (float*)   alloc((size_t)NPTS*KNB*3*4);
    float*    featb   = (float*)   alloc((size_t)NPTS*24*4);
    float*    nodebuf = (float*)   alloc((size_t)NPTS*HH*4);
    float*    ysbuf   = (float*)   alloc((size_t)BB*HH*4);
    _Float16* w1p     = (_Float16*)alloc((size_t)16*32*16*2);
    _Float16* w2p     = (_Float16*)alloc((size_t)8*16*32*16*2);
    _Float16* w3p     = (_Float16*)alloc((size_t)8*16*32*16*2);

    knn_kernel<<<NPTS, 128, 0, stream>>>(pos, idxb, offsb);
    geom_kernel<<<(NPTS + 255)/256, 256, 0, stream>>>(offsb, wsp, projb, featb);
    pack_kernel<<<(16*32*16 + 2*8*16*32*16 + 255)/256, 256, 0, stream>>>(w1, w2, w3, w1p, w2p, w3p);
    mlp_kernel<<<NPTS/4, 256, 0, stream>>>(projb, featb, idxb, w1p, b1, w2p, b2, nodebuf);
    nodemean_kernel<<<BB*HH, 256, 0, stream>>>(nodebuf, ysbuf);
    head_kernel<<<1, 256, 0, stream>>>(ysbuf, w3p, b3, w4, b4, out);
}
